// Model_24086176596770
// MI455X (gfx1250) — matrix operand order repaired and checked by reading, not compiled, not run on hardware
//
#include <hip/hip_runtime.h>
#include <hip/hip_bf16.h>
#include <cstdint>

typedef _Float16 half_t;
typedef __attribute__((ext_vector_type(16))) _Float16 v16h;
typedef __attribute__((ext_vector_type(8)))  _Float16 v8h;
typedef __attribute__((ext_vector_type(8)))  float    v8f;

// ---------------------------------------------------------------------------
// CDNA5 wave32 WMMA fragment loaders (16x16x32 f16, contiguous k = 16*(lane/16) + s, same for A and B)
// ---------------------------------------------------------------------------

// A matrix (16xK, row-major, ldk halfs per row). Per-lane:
//   row  = lane & 15
//   lanes 0-15 : K = {0..15} in v0..7   (within 32-K block)
//   lanes 16-31: K = {16..31} in v0..7
__device__ __forceinline__ v16h load_a_frag(const half_t* __restrict__ base,
                                            int ldk, int kb) {
  int lane = threadIdx.x & 31;
  int row  = lane & 15;
  int kofs = (lane >> 4) ? 16 : 0;
  const half_t* p = base + row * ldk + kb * 32 + kofs;
  v8h lo = *(const v8h*)(p);        // K block  [kofs .. kofs+7]
  v8h hi = *(const v8h*)(p + 8);    // K block  [kofs+8 .. kofs+15]
  v16h r;
  #pragma unroll
  for (int i = 0; i < 8; ++i) { r[i] = lo[i]; r[8 + i] = hi[i]; }
  return r;
}

// B matrix (32x16, column-major storage: Bc[n*128 + k], 128 = K stride).
// Per-lane: col = lane & 15; lanes 0-15 hold K=0..15, lanes 16-31 K=16..31.
__device__ __forceinline__ v16h load_b_frag(const half_t* __restrict__ colmaj,
                                            int kb) {
  int lane = threadIdx.x & 31;
  int col  = lane & 15;
  int kofs = (lane >> 4) ? 16 : 0;
  return *(const v16h*)(colmaj + col * 128 + kb * 32 + kofs);
}

// ---------------------------------------------------------------------------
// Convert a [128 x N] row-major fp32 matrix into [N x 128] column-major f16
// (B-operand layout for the scan kernel; only 96 KB per U matrix).
// ---------------------------------------------------------------------------
__global__ void conv_colmajor_f16_kernel(const float* __restrict__ W,
                                         half_t* __restrict__ Wh, int N) {
  int i = blockIdx.x * blockDim.x + threadIdx.x;
  if (i >= 128 * N) return;
  int k = i / N, n = i - k * N;
  Wh[n * 128 + k] = (half_t)W[i];
}

// ---------------------------------------------------------------------------
// XP[M,384] = X[M,128] @ W[128,384] + bias[384]   (f16 WMMA, f32 accumulate)
// grid = (M/128, 3), block = 256 (8 waves). Each wave: 16-row strip x 128 N.
// ---------------------------------------------------------------------------
__global__ __launch_bounds__(256) void gemm_xw_kernel(
    const float* __restrict__ X, const float* __restrict__ W,
    const float* __restrict__ bias, float* __restrict__ XP, int N) {
  __shared__ half_t Xs[128 * 128];  // row-major [m][k]
  __shared__ half_t Ws[128 * 128];  // col-major [n][k]
  const int m0 = blockIdx.x * 128;
  const int n0 = blockIdx.y * 128;
  const int t  = threadIdx.x;

  // Stage X tile (coalesced fp32 read -> f16 LDS)
  for (int i = t; i < 128 * 128; i += 256) {
    int r = i >> 7, c = i & 127;
    Xs[i] = (half_t)X[(size_t)(m0 + r) * 128 + c];
  }
  // Stage W tile transposed into column-major (coalesced over n)
  for (int i = t; i < 128 * 128; i += 256) {
    int k = i >> 7, n = i & 127;
    Ws[n * 128 + k] = (half_t)W[(size_t)k * N + (n0 + n)];
  }
  __syncthreads();

  const int wave = t >> 5;
  const half_t* arow = Xs + wave * 16 * 128;

  v8f acc[8] = {};
  #pragma unroll
  for (int kb = 0; kb < 4; ++kb) {
    v16h a = load_a_frag(arow, 128, kb);
    #pragma unroll
    for (int nt = 0; nt < 8; ++nt) {
      v16h b = load_b_frag(Ws + nt * 16 * 128, kb);
      acc[nt] = __builtin_amdgcn_wmma_f32_16x16x32_f16(
          false, a, false, b, (short)0, acc[nt], false, false);
    }
  }

  // Store C + bias. Lane l: col = l&15, rows = (l>=16?8:0)+v.
  const int lane = t & 31;
  const int coln = lane & 15;
  const int rofs = (lane >> 4) ? 8 : 0;
  #pragma unroll
  for (int nt = 0; nt < 8; ++nt) {
    int gn = n0 + nt * 16 + coln;
    float bv = bias[gn];
    #pragma unroll
    for (int v = 0; v < 8; ++v) {
      int gm = m0 + wave * 16 + rofs + v;
      XP[(size_t)gm * N + gn] = acc[nt][v] + bv;
    }
  }
}

// ---------------------------------------------------------------------------
// Persistent GRU scan. One workgroup per 16 batch rows (batch-parallel; the
// recurrence has no cross-batch coupling). 8 waves each own 3 of the 24
// 16x16 hp tiles. h kept in LDS (f32 state + f16 WMMA operand).
//   hp = h @ U + b[1];  z=sig(xz+hz); r=sig(xr+hr); hc=tanh(xh+r*hh)
//   h' = z*h + (1-z)*hc
// ---------------------------------------------------------------------------
__global__ __launch_bounds__(256) void gru_scan_kernel(
    const float* __restrict__ XP,   // [B, T, 384]  (x@W + b[0], precomputed)
    const half_t* __restrict__ Uh,  // [384][128] column-major f16
    const float* __restrict__ bias, // b[1], 384 values
    float* __restrict__ SEQ,        // [B, T, 128] or nullptr
    float* __restrict__ HOUT,       // [B, 128] final state or nullptr
    int T, int write_seq) {
  __shared__ half_t Hh[16 * 128];   // h as f16 (WMMA A operand)
  __shared__ float  Hf[16 * 128];   // h as f32 (state)
  __shared__ float  HP[16 * 384];   // hp accumulators

  const int b0   = blockIdx.x * 16;
  const int t    = threadIdx.x;
  const int wave = t >> 5;
  const int lane = t & 31;
  const int coln = lane & 15;
  const int rofs = (lane >> 4) ? 8 : 0;

  for (int i = t; i < 16 * 128; i += 256) { Hf[i] = 0.f; Hh[i] = (half_t)0.f; }
  __syncthreads();

  for (int step = 0; step < T; ++step) {
    // Phase 1: hp = h @ U   (all waves share the same 16x128 A operand)
    #pragma unroll
    for (int j = 0; j < 3; ++j) {
      const int nt = wave * 3 + j;
      v8f acc = {};
      #pragma unroll
      for (int kb = 0; kb < 4; ++kb) {
        v16h a = load_a_frag(Hh, 128, kb);
        v16h b = load_b_frag(Uh + nt * 16 * 128, kb);
        acc = __builtin_amdgcn_wmma_f32_16x16x32_f16(
            false, a, false, b, (short)0, acc, false, false);
      }
      const int gc = nt * 16 + coln;
      #pragma unroll
      for (int v = 0; v < 8; ++v) HP[(rofs + v) * 384 + gc] = acc[v];
    }
    __syncthreads();

    // Phase 2: elementwise GRU update of the 16x128 state
    for (int i = t; i < 16 * 128; i += 256) {
      const int r = i >> 7, c = i & 127;
      const float* xrow = XP + ((size_t)(b0 + r) * T + step) * 384;
      float xz = xrow[c], xr = xrow[128 + c], xh = xrow[256 + c];
      float hz = HP[r * 384 + c]         + bias[c];
      float hr = HP[r * 384 + 128 + c]   + bias[128 + c];
      float hh = HP[r * 384 + 256 + c]   + bias[256 + c];
      float z  = 1.f / (1.f + __expf(-(xz + hz)));
      float rr = 1.f / (1.f + __expf(-(xr + hr)));
      float hc = tanhf(xh + rr * hh);
      float hn = z * Hf[i] + (1.f - z) * hc;
      Hf[i] = hn;
      Hh[i] = (half_t)hn;
      if (write_seq)
        SEQ[((size_t)(b0 + r) * T + step) * 128 + c] = hn;
      if (step + 1 < T) __builtin_prefetch(xrow + 384, 0, 0);  // next timestep
    }
    __syncthreads();
  }

  if (HOUT) {
    for (int i = t; i < 16 * 128; i += 256) {
      int r = i >> 7, c = i & 127;
      HOUT[(size_t)(b0 + r) * 128 + c] = Hf[i];
    }
  }
}

// ---------------------------------------------------------------------------
// Head: out = sigmoid(relu(h2 @ Wd1 + bd1) @ Wd2 + bd2), one thread per row.
// ---------------------------------------------------------------------------
__global__ void head_kernel(const float* __restrict__ H2,
                            const float* __restrict__ Wd1,
                            const float* __restrict__ bd1,
                            const float* __restrict__ Wd2,
                            const float* __restrict__ bd2,
                            float* __restrict__ OUT, int B) {
  int b = blockIdx.x * blockDim.x + threadIdx.x;
  if (b >= B) return;
  const float* h = H2 + (size_t)b * 128;
  float d[64];
  #pragma unroll
  for (int j = 0; j < 64; ++j) d[j] = bd1[j];
  for (int k = 0; k < 128; ++k) {
    float hv = h[k];
    #pragma unroll
    for (int j = 0; j < 64; ++j) d[j] += hv * Wd1[k * 64 + j];
  }
  float acc = bd2[0];
  #pragma unroll
  for (int j = 0; j < 64; ++j) {
    float dv = d[j] > 0.f ? d[j] : 0.f;
    acc += dv * Wd2[j];
  }
  OUT[b] = 1.f / (1.f + __expf(-acc));
}

// ---------------------------------------------------------------------------
extern "C" void kernel_launch(void* const* d_in, const int* in_sizes, int n_in,
                              void* d_out, int out_size, void* d_ws, size_t ws_size,
                              hipStream_t stream) {
  constexpr int B = 256, T = 1024, H = 128;
  constexpr int M = B * T;       // 262144 rows for the projection GEMMs

  const float* texts = (const float*)d_in[0];
  const float* W1    = (const float*)d_in[1];
  const float* U1    = (const float*)d_in[2];
  const float* b1    = (const float*)d_in[3];
  const float* W2    = (const float*)d_in[4];
  const float* U2    = (const float*)d_in[5];
  const float* b2    = (const float*)d_in[6];
  const float* Wd1   = (const float*)d_in[7];
  const float* bd1   = (const float*)d_in[8];
  const float* Wd2   = (const float*)d_in[9];
  const float* bd2   = (const float*)d_in[10];
  float* out = (float*)d_out;

  // Workspace carve-up (256-byte aligned). xp buffer is reused by layer 2.
  auto align256 = [](size_t x) { return (x + 255) & ~(size_t)255; };
  char* ws = (char*)d_ws;
  size_t off = 0;
  float*  xp   = (float*)(ws + off); off = align256(off + (size_t)M * 384 * 4);
  float*  seq1 = (float*)(ws + off); off = align256(off + (size_t)M * H * 4);
  half_t* U1h  = (half_t*)(ws + off); off = align256(off + (size_t)384 * 128 * 2);
  half_t* U2h  = (half_t*)(ws + off); off = align256(off + (size_t)384 * 128 * 2);
  float*  h2   = (float*)(ws + off);  off = align256(off + (size_t)B * H * 4);
  (void)ws_size; (void)in_sizes; (void)n_in; (void)out_size;

  // Pre-convert recurrent weights to column-major f16 (B-operand layout).
  conv_colmajor_f16_kernel<<<(128 * 384 + 255) / 256, 256, 0, stream>>>(U1, U1h, 384);
  conv_colmajor_f16_kernel<<<(128 * 384 + 255) / 256, 256, 0, stream>>>(U2, U2h, 384);

  // Layer 1: xp = texts @ W1 + b1[0];  seq1 = scan(xp, U1, b1[1])
  gemm_xw_kernel<<<dim3(M / 128, 3), 256, 0, stream>>>(texts, W1, b1, xp, 384);
  gru_scan_kernel<<<B / 16, 256, 0, stream>>>(xp, U1h, b1 + 384, seq1, nullptr, T, 1);

  // Layer 2: xp = seq1 @ W2 + b2[0];  h2 = last state of scan(xp, U2, b2[1])
  gemm_xw_kernel<<<dim3(M / 128, 3), 256, 0, stream>>>(seq1, W2, b2, xp, 384);
  gru_scan_kernel<<<B / 16, 256, 0, stream>>>(xp, U2h, b2 + 384, nullptr, h2, T, 0);

  // Head
  head_kernel<<<1, 256, 0, stream>>>(h2, Wd1, bd1, Wd2, bd2, out, B);
}
